// PointNet_50122268344667
// MI455X (gfx1250) — matrix operand order repaired and checked by reading, not compiled, not run on hardware
//
#include <hip/hip_runtime.h>

// ---------------------------------------------------------------------------
// PointNet-style pipeline for MI455X (gfx1250, wave32, WMMA).
// Dominant cost: mlp1 over 1M points (~1.5e11 FLOPs) with a fused per-channel
// argmax so the 2.15 GB activation tensor is never materialized.
// v_wmma_f32_16x16x32_f16 with f32 accumulation; all weights LDS-resident
// (300 KB dynamic LDS -> CDNA5's 320 KB WGP LDS). Each wave owns 32 rows so
// every B fragment fetched from LDS feeds 2 WMMAs (halves DS traffic/WMMA).
// Wave-private staging + in-order LDS lets us replace per-phase s_barrier with
// wave-local ordering so SiLU TRANS ops of one wave co-execute with another
// wave's WMMA burst.
// ---------------------------------------------------------------------------

#define C_CASES   16
#define NPTS      65536
#define F1        512
#define CHUNKS    64
#define CHUNK_PTS (NPTS / CHUNKS)   // 1024 points per chunk
#define TILE_M    256               // 8 waves x 32 rows
#define NWAVES    8
#define CHUNKS_PER_WG 2             // quasi-persistent WGs: amortize LDS preload
#define LDH0      72                // padded f16 stride, 64-wide rows (16B aligned, conflict-free)
#define LDH1      136               // padded f16 stride, 128-wide rows

typedef __attribute__((ext_vector_type(16))) _Float16 v16h;
typedef __attribute__((ext_vector_type(8)))  _Float16 v8h;
typedef __attribute__((ext_vector_type(8)))  float    v8f;

// dynamic-LDS layout (bytes)
#define SM_W1T   0          // 18432
#define SM_W2T   18432      // 139264
#define SM_H0    157696     // 8*32*72*2  = 36864
#define SM_H1    194560     // 8*32*136*2 = 69632
#define SM_WMAX  264192     // 16384
// main kernel extras
#define SM_WIDX  280576     // 16384
#define SM_W0_M  296960     // 768
#define SM_B0_M  297728     // 256
#define SM_B1_M  297984     // 512
#define SM_B2_M  298496     // 2048
#define SMEM_MAIN 300544
// head kernel extras
#define SM_LAT0  280576     // 2048
#define SM_HBUF  282624     // 1024
#define SM_W0_H  283648     // 768
#define SM_B0_H  284416     // 256
#define SM_B1_H  284672     // 512
#define SM_B2_H  285184     // 2048
#define SMEM_HEAD 287232

__device__ __forceinline__ float silu_f(float x) {
  // x * sigmoid(x); v_exp_f32 + v_rcp_f32 are TRANS ops -> co-execute with WMMA
  return x * __builtin_amdgcn_rcpf(1.0f + __expf(-x));
}

__device__ __forceinline__ v16h cat8(v8h lo, v8h hi) {
  return __builtin_shufflevector(lo, hi, 0,1,2,3,4,5,6,7,8,9,10,11,12,13,14,15);
}

// A fragment (16x32 f16, MxK) from row-major LDS stage (stride ldh halves).
// ISA 7.12.2: lanes 0-15 row M=lane hold K=k0+[0..7] and k0+16+[0..7];
// lanes 16-31 hold K=k0+8+[0..7] and k0+24+[0..7].
__device__ __forceinline__ v16h load_a(const _Float16* base, int ldh, int lane, int k0) {
  int m  = lane & 15;
  int kb = k0 + ((lane >> 4) << 3);
  const _Float16* p = base + m * ldh + kb;
  return cat8(*(const v8h*)p, *(const v8h*)(p + 16));
}

// B fragment (32x16 f16, KxN) from N-major (transposed) weights, stride ldh.
// lanes 0-15: col N=n0+lane, K=k0+[0..7] and k0+16+[0..7]; lanes 16-31: K=k0+8+[0..7] and k0+24+[0..7].
__device__ __forceinline__ v16h load_b(const _Float16* wt, int ldh, int lane, int n0, int k0) {
  int n  = n0 + (lane & 15);
  int kb = k0 + ((lane >> 4) << 3);
  const _Float16* p = wt + n * ldh + kb;
  return cat8(*(const v8h*)p, *(const v8h*)(p + 16));
}

// wave-local ordering between LDS phases: same-wave DS ops are in-order per
// ISA, we only need to stop compiler reordering. No s_barrier -> waves drift
// and overlap VALU/TRANS epilogues with other waves' WMMA.
__device__ __forceinline__ void wave_phase() { __builtin_amdgcn_wave_barrier(); }

__device__ __forceinline__ void preload_weights(
    int t,
    const float* W1_0, const float* b1_0, const float* b1_1, const float* b1_2,
    const _Float16* W1_1t, const _Float16* W1_2t,
    float* sW0, float* sb0, float* sB1, float* sB2,
    _Float16* sW1t, _Float16* sW2t)
{
  for (int i = t; i < 1024; i += 256) {            // W1_1^T: 128 x 64 halves
    int n = i >> 3, k = (i & 7) << 3;
    *(v8h*)&sW1t[n * LDH0 + k] = *(const v8h*)&W1_1t[n * 64 + k];
  }
  for (int i = t; i < 8192; i += 256) {            // W1_2^T: 512 x 128 halves
    int n = i >> 4, k = (i & 15) << 3;
    *(v8h*)&sW2t[n * LDH1 + k] = *(const v8h*)&W1_2t[n * 128 + k];
  }
  if (t < 192) sW0[t] = W1_0[t];
  if (t < 64)  sb0[t] = b1_0[t];
  if (t < 128) sB1[t] = b1_1[t];
  for (int i = t; i < F1; i += 256) sB2[i] = b1_2[i];
}

// One 256-point tile through mlp1 with fused bias+SiLU and per-channel
// (arg)max epilogue into per-wave LDS slots. All staging is wave-private.
template <bool TRACK_IDX>
__device__ __forceinline__ void mlp1_tile(
    const float* ptile, int idx_base, bool prefetch_next,
    int t, int lane, int w,
    const float* sW0, const float* sb0, const float* sB1, const float* sB2,
    const _Float16* sW1t, const _Float16* sW2t,
    _Float16* h0s, _Float16* h1s, float* wmax, int* widx)
{
  // ---- layer 1 (K=3, VALU): thread t -> row t, all 64 channels ----
  {
    if (prefetch_next)
      __builtin_prefetch(ptile + TILE_M * 3 + t * 3, 0, 0);   // global_prefetch_b8
    const float* p = ptile + t * 3;
    float px = p[0], py = p[1], pz = p[2];
    _Float16* dst = h0s + w * (32 * LDH0) + (t & 31) * LDH0;
    #pragma unroll
    for (int j = 0; j < 64; ++j) {
      float xv = fmaf(px, sW0[j], fmaf(py, sW0[64 + j], fmaf(pz, sW0[128 + j], sb0[j])));
      dst[j] = (_Float16)silu_f(xv);
    }
  }
  wave_phase();

  // ---- layer 2: per wave 32x128, K=64 -> 32 WMMA (B reused across 2 M-tiles) ----
  {
    const _Float16* aw = h0s + w * (32 * LDH0);
    v16h a[2][2];
    #pragma unroll
    for (int mt = 0; mt < 2; ++mt)
      #pragma unroll
      for (int kt = 0; kt < 2; ++kt)
        a[mt][kt] = load_a(aw + mt * 16 * LDH0, LDH0, lane, kt << 5);
    _Float16* hw = h1s + w * (32 * LDH1);
    int nn_lo = lane & 15;
    int mhi   = (lane >> 4) << 3;
    #pragma unroll
    for (int nt = 0; nt < 8; ++nt) {
      v16h bf0 = load_b(sW1t, LDH0, lane, nt << 4, 0);
      v16h bf1 = load_b(sW1t, LDH0, lane, nt << 4, 32);
      float bias = sB1[(nt << 4) + nn_lo];
      int nn = (nt << 4) + nn_lo;
      #pragma unroll
      for (int mt = 0; mt < 2; ++mt) {
        v8f acc;
        #pragma unroll
        for (int i = 0; i < 8; ++i) acc[i] = bias;
        acc = __builtin_amdgcn_wmma_f32_16x16x32_f16(false, a[mt][0], false, bf0, (short)0, acc, false, false);
        acc = __builtin_amdgcn_wmma_f32_16x16x32_f16(false, a[mt][1], false, bf1, (short)0, acc, false, false);
        #pragma unroll
        for (int r = 0; r < 8; ++r)
          hw[(mt * 16 + mhi + r) * LDH1 + nn] = (_Float16)silu_f(acc[r]);  // D-layout scatter
      }
    }
  }
  wave_phase();

  // ---- layer 3: per wave 32x512, K=128 -> 256 WMMA + fused channel (arg)max ----
  {
    const _Float16* aw = h1s + w * (32 * LDH1);
    v16h a[2][4];
    #pragma unroll
    for (int mt = 0; mt < 2; ++mt)
      #pragma unroll
      for (int kt = 0; kt < 4; ++kt)
        a[mt][kt] = load_a(aw + mt * 16 * LDH1, LDH1, lane, kt << 5);
    int nn_lo = lane & 15;
    int mb0 = idx_base + (w << 5) + ((lane >> 4) << 3);   // point id of acc[0], mt=0
    #pragma unroll 2
    for (int nt = 0; nt < 32; ++nt) {
      v16h bf[4];
      #pragma unroll
      for (int kt = 0; kt < 4; ++kt)
        bf[kt] = load_b(sW2t, LDH1, lane, nt << 4, kt << 5);
      float bias = sB2[(nt << 4) + nn_lo];
      #pragma unroll
      for (int mt = 0; mt < 2; ++mt) {
        v8f acc;
        #pragma unroll
        for (int i = 0; i < 8; ++i) acc[i] = bias;
        #pragma unroll
        for (int kt = 0; kt < 4; ++kt)
          acc = __builtin_amdgcn_wmma_f32_16x16x32_f16(false, a[mt][kt], false, bf[kt], (short)0, acc, false, false);
        float bv = -3.4e38f; int bi = 0;
        int mb = mb0 + mt * 16;
        #pragma unroll
        for (int r = 0; r < 8; ++r) {                     // this lane's 8 M-rows
          float y = silu_f(acc[r]);
          if (y > bv) { bv = y; bi = mb + r; }
        }
        float ov = __shfl_xor(bv, 16, 32);                // merge halves (M 0-7 vs 8-15)
        int   oi = __shfl_xor(bi, 16, 32);
        if (ov > bv) { bv = ov; bi = oi; }
        if (lane < 16) {
          int ch = (nt << 4) + lane;
          float* pm = wmax + w * F1 + ch;                 // wave-private slot
          if (bv > *pm) {
            *pm = bv;
            if (TRACK_IDX) widx[w * F1 + ch] = bi;
          }
        }
      }
    }
  }
  wave_phase();
}

// ---------------------------------------------------------------------------
__global__ void convert_weights(const float* __restrict__ W1_1,
                                const float* __restrict__ W1_2,
                                _Float16* __restrict__ W1_1t,
                                _Float16* __restrict__ W1_2t)
{
  int i = blockIdx.x * 256 + threadIdx.x;
  if (i < 8192) {                      // [64][128] -> [128][64]
    int n = i >> 6, k = i & 63;
    W1_1t[i] = (_Float16)W1_1[k * 128 + n];
  } else if (i < 73728) {              // [128][512] -> [512][128]
    int j = i - 8192;
    int n = j >> 7, k = j & 127;
    W1_2t[j] = (_Float16)W1_2[k * 512 + n];
  }
}

__launch_bounds__(256, 1)
__global__ void mlp_argmax_kernel(const float* __restrict__ pts,
    const float* __restrict__ W1_0, const float* __restrict__ b1_0,
    const _Float16* __restrict__ W1_1t, const float* __restrict__ b1_1,
    const _Float16* __restrict__ W1_2t, const float* __restrict__ b1_2,
    float* __restrict__ pmax, int* __restrict__ pidx)
{
  extern __shared__ __align__(16) char smem[];
  _Float16* sW1t = (_Float16*)(smem + SM_W1T);
  _Float16* sW2t = (_Float16*)(smem + SM_W2T);
  _Float16* h0s  = (_Float16*)(smem + SM_H0);
  _Float16* h1s  = (_Float16*)(smem + SM_H1);
  float* wmax = (float*)(smem + SM_WMAX);
  int*   widx = (int*)  (smem + SM_WIDX);
  float* sW0  = (float*)(smem + SM_W0_M);
  float* sb0  = (float*)(smem + SM_B0_M);
  float* sB1  = (float*)(smem + SM_B1_M);
  float* sB2  = (float*)(smem + SM_B2_M);

  const int t = threadIdx.x, lane = t & 31, w = t >> 5;

  preload_weights(t, W1_0, b1_0, b1_1, b1_2, W1_1t, W1_2t,
                  sW0, sb0, sB1, sB2, sW1t, sW2t);
  __syncthreads();

  // quasi-persistent: each WG processes CHUNKS_PER_WG chunks with one preload
  for (int cc = 0; cc < CHUNKS_PER_WG; ++cc) {
    const int flat  = blockIdx.x * CHUNKS_PER_WG + cc;   // [0, C*CHUNKS)
    const int c     = flat >> 6;
    const int chunk = flat & 63;

    for (int i = t; i < NWAVES * F1; i += 256) { wmax[i] = -3.4e38f; widx[i] = 0; }
    __syncthreads();

    const float* cpts = pts + (size_t)c * NPTS * 3;
    const int base = chunk * CHUNK_PTS;
    #pragma unroll 1
    for (int tile = 0; tile < CHUNK_PTS / TILE_M; ++tile) {
      int tb = base + tile * TILE_M;
      mlp1_tile<true>(cpts + (size_t)tb * 3, tb, tile + 1 < CHUNK_PTS / TILE_M,
                      t, lane, w, sW0, sb0, sB1, sB2, sW1t, sW2t,
                      h0s, h1s, wmax, widx);
    }
    __syncthreads();

    // merge 8 wave slots -> per-chunk partial (max, argidx)
    for (int ch = t; ch < F1; ch += 256) {
      float bv = wmax[ch]; int bi = widx[ch];
      #pragma unroll
      for (int ww = 1; ww < NWAVES; ++ww) {
        float v = wmax[ww * F1 + ch];
        if (v > bv) { bv = v; bi = widx[ww * F1 + ch]; }
      }
      size_t o = (size_t)flat * F1 + ch;
      pmax[o] = bv; pidx[o] = bi;
    }
    __syncthreads();
  }
}

__global__ void reduce_gather_kernel(const float* __restrict__ pmax,
                                     const int* __restrict__ pidx,
                                     const float* __restrict__ pts,
                                     float* __restrict__ fixedp)
{
  int c = blockIdx.x, ch = threadIdx.x;   // 512 threads
  size_t rb = (size_t)c * CHUNKS * F1 + ch;
  float bv = -3.4e38f; int bi = 0;
  for (int k = 0; k < CHUNKS; ++k) {
    float v = pmax[rb + (size_t)k * F1];
    if (v > bv) { bv = v; bi = pidx[rb + (size_t)k * F1]; }
  }
  const float* p = pts + ((size_t)c * NPTS + bi) * 3;
  float* f = fixedp + ((size_t)c * F1 + ch) * 3;
  f[0] = p[0]; f[1] = p[1]; f[2] = p[2];
}

__launch_bounds__(256, 1)
__global__ void head_kernel(const float* __restrict__ fixedp,
    const float* __restrict__ W1_0, const float* __restrict__ b1_0,
    const _Float16* __restrict__ W1_1t, const float* __restrict__ b1_1,
    const _Float16* __restrict__ W1_2t, const float* __restrict__ b1_2,
    const float* __restrict__ W2_0, const float* __restrict__ b2_0,
    const float* __restrict__ W2_1, const float* __restrict__ b2_1,
    float* __restrict__ latent)
{
  extern __shared__ __align__(16) char smem[];
  _Float16* sW1t = (_Float16*)(smem + SM_W1T);
  _Float16* sW2t = (_Float16*)(smem + SM_W2T);
  _Float16* h0s  = (_Float16*)(smem + SM_H0);
  _Float16* h1s  = (_Float16*)(smem + SM_H1);
  float* wmax = (float*)(smem + SM_WMAX);
  float* lat0 = (float*)(smem + SM_LAT0);
  float* hbuf = (float*)(smem + SM_HBUF);
  float* sW0  = (float*)(smem + SM_W0_H);
  float* sb0  = (float*)(smem + SM_B0_H);
  float* sB1  = (float*)(smem + SM_B1_H);
  float* sB2  = (float*)(smem + SM_B2_H);

  const int t = threadIdx.x, lane = t & 31, w = t >> 5;
  const int c = blockIdx.x;

  preload_weights(t, W1_0, b1_0, b1_1, b1_2, W1_1t, W1_2t,
                  sW0, sb0, sB1, sB2, sW1t, sW2t);
  for (int i = t; i < NWAVES * F1; i += 256) wmax[i] = -3.4e38f;
  __syncthreads();

  const float* cfix = fixedp + (size_t)c * F1 * 3;
  #pragma unroll 1
  for (int tile = 0; tile < F1 / TILE_M; ++tile) {
    int tb = tile * TILE_M;
    mlp1_tile<false>(cfix + (size_t)tb * 3, tb, tile + 1 < F1 / TILE_M,
                     t, lane, w, sW0, sb0, sB1, sB2, sW1t, sW2t,
                     h0s, h1s, wmax, nullptr);
  }
  __syncthreads();

  for (int ch = t; ch < F1; ch += 256) {
    float bv = wmax[ch];
    #pragma unroll
    for (int ww = 1; ww < NWAVES; ++ww) bv = fmaxf(bv, wmax[ww * F1 + ch]);
    lat0[ch] = bv;
  }
  __syncthreads();

  // h = silu(lat0 @ W2_0 + b2_0): thread t -> output channel t (of 256)
  {
    float a = b2_0[t];
    for (int k = 0; k < F1; ++k) a = fmaf(lat0[k], W2_0[k * 256 + t], a);
    hbuf[t] = silu_f(a);
  }
  __syncthreads();

  // latent = h @ W2_1 + b2_1
  for (int f = t; f < F1; f += 256) {
    float a = b2_1[f];
    for (int k = 0; k < 256; ++k) a = fmaf(hbuf[k], W2_1[k * F1 + f], a);
    latent[(size_t)c * F1 + f] = a;
  }
}

__global__ void scatter_kernel(const int* __restrict__ x,
                               const float* __restrict__ latent,
                               float* __restrict__ out)
{
  int i = blockIdx.x * 256 + threadIdx.x;   // B*512 elements
  int b = i >> 9, f = i & 511;
  out[i] = latent[(size_t)x[b] * F1 + f];
}

// ---------------------------------------------------------------------------
extern "C" void kernel_launch(void* const* d_in, const int* in_sizes, int n_in,
                              void* d_out, int out_size, void* d_ws, size_t ws_size,
                              hipStream_t stream) {
  (void)in_sizes; (void)n_in; (void)out_size; (void)ws_size;
  const float* pts  = (const float*)d_in[0];
  const float* W1_0 = (const float*)d_in[1];
  const float* b1_0 = (const float*)d_in[2];
  const float* W1_1 = (const float*)d_in[3];
  const float* b1_1 = (const float*)d_in[4];
  const float* W1_2 = (const float*)d_in[5];
  const float* b1_2 = (const float*)d_in[6];
  const float* W2_0 = (const float*)d_in[7];
  const float* b2_0 = (const float*)d_in[8];
  const float* W2_1 = (const float*)d_in[9];
  const float* b2_1 = (const float*)d_in[10];
  const int*   x    = (const int*)d_in[11];

  char* ws = (char*)d_ws;
  _Float16* W1_1t = (_Float16*)(ws);                          // 16384 B
  _Float16* W1_2t = (_Float16*)(ws + 16384);                  // 131072 B
  float* pmax  = (float*)(ws + 147456);                       // 2 MB
  int*   pidx  = (int*)  (ws + 147456 + 2097152);             // 2 MB
  float* fixedp= (float*)(ws + 147456 + 2 * 2097152);         // 98304 B
  float* latent= (float*)(ws + 147456 + 2 * 2097152 + 98304); // 32768 B

  convert_weights<<<288, 256, 0, stream>>>(W1_1, W1_2, W1_1t, W1_2t);

  mlp_argmax_kernel<<<(C_CASES * CHUNKS) / CHUNKS_PER_WG, 256, SMEM_MAIN, stream>>>(
      pts, W1_0, b1_0, W1_1t, b1_1, W1_2t, b1_2, pmax, pidx);

  reduce_gather_kernel<<<C_CASES, 512, 0, stream>>>(pmax, pidx, pts, fixedp);

  head_kernel<<<C_CASES, 256, SMEM_HEAD, stream>>>(
      fixedp, W1_0, b1_0, W1_1t, b1_1, W1_2t, b1_2, W2_0, b2_0, W2_1, b2_1, latent);

  scatter_kernel<<<(4096 * 512) / 256, 256, 0, stream>>>(x, latent, (float*)d_out);
}
